// dualGCN_53532472377641
// MI455X (gfx1250) — compile-verified, matrix operand order checked by reading, not run on hardware
//
#include <hip/hip_runtime.h>
#include <hip/hip_bf16.h>
#include <math.h>

// ---------------------------------------------------------------------------
// Dual-branch GCN for MI455X (gfx1250, wave32).
//  - Dense GEMMs: v_wmma_f32_16x16x32_bf16, A staged in LDS (float4 -> packed
//    bf16), W pre-packed to bf16 in WMMA fragment order (branchless b128 loads).
//  - Edge aggregation: float4 gathers + global_atomic_add_f32; h buffers
//    (51.2MB) are L2-resident (192MB L2), so this phase runs at L2 bandwidth.
// ---------------------------------------------------------------------------

#define NODES  50000
#define HDIM   256
#define FIN    512
#define LOUT   52
#define MAXK   512
#define LDSPAD 8   // keeps row stride a multiple of 8 elements (16B) + de-conflicts banks

typedef __attribute__((ext_vector_type(16))) __bf16 v16bf;
typedef __attribute__((ext_vector_type(8)))  float  v8f;

union V16 { v16bf v; unsigned short s[16]; uint4 q[2]; };

__device__ __forceinline__ unsigned short f2bf(float f) {
  union { float f; unsigned int u; } c; c.f = f;
  unsigned int u = c.u;
  u += 0x7FFFu + ((u >> 16) & 1u);      // round-to-nearest-even
  return (unsigned short)(u >> 16);
}

__device__ __forceinline__ int klmap(int j, int kh) {
  // 16-bit A/B fragment element -> K offset within a 32-wide k-block
  return (j < 8) ? (kh * 8 + j) : (16 + kh * 8 + (j - 8));
}

// ---------------------------------------------------------------------------
// Pack W[K x cols] (f32, row-major) -> bf16 fragment-order:
//   Wpk[(((kb*colsPad + c) * 2) + kh) * 16 + j] = bf16(W[(kb*32+klmap(j,kh))*cols + c])
// Zero-padded for c >= cols, so the GEMM hot loop is branchless.
// ---------------------------------------------------------------------------
__global__ void pack_w_bf16(const float* __restrict__ W, unsigned short* __restrict__ Wpk,
                            int K, int cols, int colsPad) {
  long long total = (long long)(K >> 5) * colsPad * 32;
  long long i = (long long)blockIdx.x * blockDim.x + threadIdx.x;
  if (i >= total) return;
  int j  = (int)(i & 15);
  int kh = (int)((i >> 4) & 1);
  long long rest = i >> 5;
  int c  = (int)(rest % colsPad);
  int kb = (int)(rest / colsPad);
  int k  = (kb << 5) + klmap(j, kh);
  float v = (c < cols) ? W[(long long)k * cols + c] : 0.0f;
  Wpk[i] = f2bf(v);
}

// ---------------------------------------------------------------------------
// C[nrows x cols] = act( A[nrows x K] @ W[K x cols] + bias )
// relu_in: ReLU fused into the A->bf16 staging. act: 0=none, 1=relu, 2=sigmoid.
// One wave per 16x16 output tile; K-loop of 16x16x32 bf16 WMMAs.
// ---------------------------------------------------------------------------
__global__ void gemm_wmma_bf16(const float* __restrict__ A,
                               const unsigned short* __restrict__ Wpk,
                               const float* __restrict__ bias,
                               float* __restrict__ C,
                               int nrows, int K, int cols,
                               int relu_in, int act) {
  __shared__ unsigned short ldsA[16 * (MAXK + LDSPAD)];
  const int tid = threadIdx.x;
  const long long rowBase = (long long)blockIdx.x * 16;
  const int ldk = K + LDSPAD;
  const int Kq  = K >> 2;

  // Stage 16 rows x K of A as bf16 in LDS: float4 loads, packed uint2 stores.
  for (int i = tid; i < 16 * Kq; i += blockDim.x) {
    int r = i / Kq, kq = i - r * Kq;
    long long row = rowBase + r;
    float4 v = make_float4(0.f, 0.f, 0.f, 0.f);
    if (row < nrows) v = ((const float4*)(A + row * (long long)K))[kq];
    if (relu_in) {
      v.x = fmaxf(v.x, 0.f); v.y = fmaxf(v.y, 0.f);
      v.z = fmaxf(v.z, 0.f); v.w = fmaxf(v.w, 0.f);
    }
    unsigned int u0 = (unsigned int)f2bf(v.x) | ((unsigned int)f2bf(v.y) << 16);
    unsigned int u1 = (unsigned int)f2bf(v.z) | ((unsigned int)f2bf(v.w) << 16);
    *(uint2*)&ldsA[r * ldk + (kq << 2)] = make_uint2(u0, u1);
  }
  __syncthreads();

  const int wave = tid >> 5;
  const int lane = tid & 31;
  const int numTiles = (cols + 15) >> 4;
  if (wave >= numTiles) return;
  const int colsPad = numTiles << 4;

  const int m    = lane & 15;        // A fragment row == C column-within-tile
  const int kh   = lane >> 4;        // K-half select (documented wave32 layout)
  const int colN = (wave << 4) + m;  // packed/global output column

  const unsigned short* arow = ldsA + m * ldk;
  v8f acc = {};
  for (int kb = 0; kb < (K >> 5); ++kb) {
    const int k0 = kb << 5;
    V16 a, b;
    // A: two contiguous 8-element (16B) runs in LDS -> ds_load_b128 x2
    a.q[0] = *(const uint4*)&arow[k0 + kh * 8];
    a.q[1] = *(const uint4*)&arow[k0 + 16 + kh * 8];
    // B: 32B contiguous per lane in packed W -> global_load_b128 x2, branchless
    const unsigned short* bp = Wpk + ((((long long)kb * colsPad + colN) << 1) + kh) * 16;
    b.q[0] = *(const uint4*)&bp[0];
    b.q[1] = *(const uint4*)&bp[8];
    acc = __builtin_amdgcn_wmma_f32_16x16x32_bf16(
        /*neg_a=*/false, a.v, /*neg_b=*/false, b.v,
        /*c_mod=*/(short)0, acc, /*reuse_a=*/false, /*reuse_b=*/false);
  }

  // C/D layout: VGPR r -> M = r + 8*(lane>=16), N = lane%16.
#pragma unroll
  for (int r = 0; r < 8; ++r) {
    long long row = rowBase + r + ((lane >= 16) ? 8 : 0);
    if (row < nrows && colN < cols) {
      float v = acc[r];
      if (bias) v += bias[colN];
      if (act == 1)      v = fmaxf(v, 0.0f);
      else if (act == 2) v = 1.0f / (1.0f + __expf(-v));
      C[row * (long long)cols + colN] = v;
    }
  }
}

// ---------------------------------------------------------------------------
// Degree / normalization kernels
// ---------------------------------------------------------------------------
__global__ void zero_u32(unsigned int* d, int n) {
  int i = blockIdx.x * blockDim.x + threadIdx.x;
  if (i < n) d[i] = 0u;
}

__global__ void count_deg(const int* __restrict__ dst, unsigned int* __restrict__ deg,
                          long long nE) {
  long long e = (long long)blockIdx.x * blockDim.x + threadIdx.x;
  if (e < nE) atomicAdd(&deg[dst[e]], 1u);
}

__global__ void make_dinv(const unsigned int* __restrict__ deg, float* __restrict__ dinv,
                          int n) {
  int i = blockIdx.x * blockDim.x + threadIdx.x;
  if (i < n) dinv[i] = rsqrtf((float)deg[i] + 1.0f);  // +1 for self-loop
}

// out[n][f] = dinv[n]^2 * h[n][f] + bias[f]   (self-loop term + deferred bias)
__global__ void self_init(const float* __restrict__ h, const float* __restrict__ dinv,
                          const float* __restrict__ bias, float* __restrict__ out,
                          int n) {
  long long i = (long long)blockIdx.x * blockDim.x + threadIdx.x;   // float4 index
  if (i >= (long long)n * (HDIM / 4)) return;
  int node = (int)(i >> 6);
  int f4   = (int)(i & 63);
  float di = dinv[node];
  float s  = di * di;
  float4 h4 = ((const float4*)h)[i];
  const float4 b4 = ((const float4*)bias)[f4];
  float4 o;
  o.x = s * h4.x + b4.x; o.y = s * h4.y + b4.y;
  o.z = s * h4.z + b4.z; o.w = s * h4.w + b4.w;
  ((float4*)out)[i] = o;
}

// out[dst] += dinv[src]*dinv[dst] * h[src] ; 64 lanes/edge x float4 each.
__global__ void edge_agg(const int* __restrict__ src, const int* __restrict__ dst,
                         const float* __restrict__ dinv, const float* __restrict__ h,
                         float* __restrict__ out, long long nE) {
  long long t = (long long)blockIdx.x * blockDim.x + threadIdx.x;
  long long e = t >> 6;
  if (e >= nE) return;
  int fi = (int)(t & 63);
  int s = src[e], d = dst[e];
  float w = dinv[s] * dinv[d];
  const float4 v = ((const float4*)(h + (long long)s * HDIM))[fi];
  float* o = out + (long long)d * HDIM + fi * 4;
  atomicAdd(o + 0, w * v.x);
  atomicAdd(o + 1, w * v.y);
  atomicAdd(o + 2, w * v.z);
  atomicAdd(o + 3, w * v.w);
}

// ---------------------------------------------------------------------------
// Gating: one wave per node. f = sigmoid(h.w + b) per branch, blend.
// ---------------------------------------------------------------------------
__global__ void gate_blend(const float* __restrict__ h1, const float* __restrict__ h2,
                           const float* __restrict__ w1W, const float* __restrict__ w1b,
                           const float* __restrict__ w2W, const float* __restrict__ w2b,
                           float* __restrict__ blend, int n) {
  int wavesPerBlock = blockDim.x >> 5;
  int node = blockIdx.x * wavesPerBlock + (threadIdx.x >> 5);
  int lane = threadIdx.x & 31;
  if (node >= n) return;
  const float* p1 = h1 + (long long)node * HDIM;
  const float* p2 = h2 + (long long)node * HDIM;
  float d1 = 0.0f, d2 = 0.0f;
  float v1[8], v2[8];
#pragma unroll
  for (int j = 0; j < 8; ++j) {
    int f = j * 32 + lane;                 // coalesced
    v1[j] = p1[f]; v2[j] = p2[f];
    d1 += v1[j] * w1W[f];
    d2 += v2[j] * w2W[f];
  }
#pragma unroll
  for (int off = 16; off >= 1; off >>= 1) {
    d1 += __shfl_xor(d1, off, 32);
    d2 += __shfl_xor(d2, off, 32);
  }
  float f1 = 1.0f / (1.0f + __expf(-(d1 + w1b[0])));
  float f2 = 1.0f / (1.0f + __expf(-(d2 + w2b[0])));
  float g  = f1 / (f1 + f2);
  float* ob = blend + (long long)node * HDIM;
#pragma unroll
  for (int j = 0; j < 8; ++j) {
    int f = j * 32 + lane;
    ob[f] = g * v1[j] + (1.0f - g) * v2[j];
  }
}

// ---------------------------------------------------------------------------
// Host-side orchestration
// ---------------------------------------------------------------------------
static void run_gemm(const float* A, const float* W, const float* bias, float* C,
                     int K, int cols, int relu_in, int act,
                     unsigned short* d_wpk, hipStream_t stream) {
  const int colsPad = (cols + 15) & ~15;
  const long long pk = (long long)(K >> 5) * colsPad * 32;
  pack_w_bf16<<<(unsigned)((pk + 255) / 256), 256, 0, stream>>>(W, d_wpk, K, cols, colsPad);
  dim3 gGemm((NODES + 15) / 16);
  gemm_wmma_bf16<<<gGemm, 32 * (colsPad / 16), 0, stream>>>(
      A, d_wpk, bias, C, NODES, K, cols, relu_in, act);
}

static void run_gcn_layer(const float* A, int K, const float* W, const float* b,
                          const int* src, const int* dst, long long nE,
                          const float* dinv, float* h_tmp, float* out,
                          int relu_in, unsigned short* d_wpk, hipStream_t stream) {
  run_gemm(A, W, nullptr, h_tmp, K, HDIM, relu_in, /*act=*/0, d_wpk, stream);
  long long nf4 = (long long)NODES * (HDIM / 4);
  self_init<<<(unsigned)((nf4 + 255) / 256), 256, 0, stream>>>(h_tmp, dinv, b, out, NODES);
  long long tthr = nE * 64;
  edge_agg<<<(unsigned)((tthr + 255) / 256), 256, 0, stream>>>(src, dst, dinv, h_tmp, out, nE);
}

static void run_branch(const float* x, const int* ei, long long nE,
                       const float* W1, const float* b1, const float* W2, const float* b2,
                       const float* W3, const float* b3, const float* We, const float* be,
                       float* d_tmp, float* d_cur, float* d_hout,
                       unsigned int* d_deg, float* d_dinv, unsigned short* d_wpk,
                       hipStream_t stream) {
  const int* src = ei;
  const int* dst = ei + nE;
  zero_u32<<<(NODES + 255) / 256, 256, 0, stream>>>(d_deg, NODES);
  count_deg<<<(unsigned)((nE + 255) / 256), 256, 0, stream>>>(dst, d_deg, nE);
  make_dinv<<<(NODES + 255) / 256, 256, 0, stream>>>(d_deg, d_dinv, NODES);

  run_gcn_layer(x,     FIN,  W1, b1, src, dst, nE, d_dinv, d_tmp, d_cur,  0, d_wpk, stream);
  run_gcn_layer(d_cur, HDIM, W2, b2, src, dst, nE, d_dinv, d_tmp, d_cur,  1, d_wpk, stream);
  run_gcn_layer(d_cur, HDIM, W3, b3, src, dst, nE, d_dinv, d_tmp, d_cur,  1, d_wpk, stream);
  run_gcn_layer(d_cur, HDIM, We, be, src, dst, nE, d_dinv, d_tmp, d_hout, 1, d_wpk, stream);
}

extern "C" void kernel_launch(void* const* d_in, const int* in_sizes, int n_in,
                              void* d_out, int out_size, void* d_ws, size_t ws_size,
                              hipStream_t stream) {
  // setup_inputs() dict order
  const float* x1   = (const float*)d_in[0];
  const float* x2   = (const float*)d_in[1];
  const int*   ei1  = (const int*)d_in[2];
  const int*   ei2  = (const int*)d_in[3];
  const float* Wc11 = (const float*)d_in[4];  const float* bc11 = (const float*)d_in[5];
  const float* Wc21 = (const float*)d_in[6];  const float* bc21 = (const float*)d_in[7];
  const float* Wc31 = (const float*)d_in[8];  const float* bc31 = (const float*)d_in[9];
  const float* Wce1 = (const float*)d_in[10]; const float* bce1 = (const float*)d_in[11];
  const float* Wc12 = (const float*)d_in[12]; const float* bc12 = (const float*)d_in[13];
  const float* Wc22 = (const float*)d_in[14]; const float* bc22 = (const float*)d_in[15];
  const float* Wc32 = (const float*)d_in[16]; const float* bc32 = (const float*)d_in[17];
  const float* Wce2 = (const float*)d_in[18]; const float* bce2 = (const float*)d_in[19];
  const float* w1W  = (const float*)d_in[20]; const float* w1b  = (const float*)d_in[21];
  const float* w2W  = (const float*)d_in[22]; const float* w2b  = (const float*)d_in[23];
  const float* finW = (const float*)d_in[24]; const float* finb = (const float*)d_in[25];
  const float* outW = (const float*)d_in[26]; const float* outb = (const float*)d_in[27];

  const long long nE = (long long)in_sizes[2] / 2;

  // Workspace carve-up
  char* p = (char*)d_ws;
  const size_t nhBytes = (size_t)NODES * HDIM * sizeof(float);
  float*          d_tmp  = (float*)p;          p += nhBytes;   // GEMM outputs / blend
  float*          d_cur  = (float*)p;          p += nhBytes;   // layer activations
  float*          d_h1   = (float*)p;          p += nhBytes;   // branch 1 result
  float*          d_h2   = (float*)p;          p += nhBytes;   // branch 2 result
  unsigned int*   d_deg  = (unsigned int*)p;   p += (size_t)NODES * sizeof(unsigned int);
  float*          d_dinv = (float*)p;          p += (size_t)NODES * sizeof(float);
  unsigned short* d_wpk  = (unsigned short*)p; p += (size_t)(FIN / 32) * HDIM * 32 * sizeof(unsigned short);
  (void)ws_size; (void)n_in; (void)out_size;

  run_branch(x1, ei1, nE, Wc11, bc11, Wc21, bc21, Wc31, bc31, Wce1, bce1,
             d_tmp, d_cur, d_h1, d_deg, d_dinv, d_wpk, stream);
  run_branch(x2, ei2, nE, Wc12, bc12, Wc22, bc22, Wc32, bc32, Wce2, bce2,
             d_tmp, d_cur, d_h2, d_deg, d_dinv, d_wpk, stream);

  // Gating -> blend in d_tmp
  gate_blend<<<(NODES * 32 + 255) / 256, 256, 0, stream>>>(
      d_h1, d_h2, w1W, w1b, w2W, w2b, d_tmp, NODES);

  // fin: relu(blend @ finW + finb) -> d_cur
  run_gemm(d_tmp, finW, finb, d_cur, HDIM, HDIM, /*relu_in=*/0, /*act=*/1, d_wpk, stream);
  // out: sigmoid(d_cur @ outW + outb) -> d_out (52 cols -> 4 col tiles)
  run_gemm(d_cur, outW, outb, (float*)d_out, HDIM, LOUT, /*relu_in=*/0, /*act=*/2, d_wpk, stream);
}
